// CausalSelfAttention_33260226740560
// MI455X (gfx1250) — matrix operand order repaired and checked by reading, not compiled, not run on hardware
//
// MI455X (gfx1250) fused sliding-window causal self-attention.
// ~43 GFLOP vs ~50MB unique HBM traffic -> matrix-core bound: everything runs
// through v_wmma_f32_16x16x32_bf16 with f32 accumulation.
// Attention is fully TRANSPOSED (S^T = K*Q^T, O^T = V^T*P^T): query lands on
// the WMMA lane axis so online-softmax stats are per-lane scalars. The key
// loop is split into {window-masked, unmasked body, causal diagonal} so only
// 2 of ~17 blocks pay mask VALU. Softmax runs in base-2; the 0.125*log2e
// scale is folded into the q-columns of w_attn at transpose time. Q/K are
// read directly from the qkv buffer (WMMA frags only need 16B contiguity);
// only V^T is materialized. GEMMs double-buffer fragments in registers to
// overlap global_load_b128 bursts with the 8-deep wmma chain.
#include <hip/hip_runtime.h>
#include <math.h>

typedef __attribute__((ext_vector_type(16))) __bf16       v16bf;
typedef __attribute__((ext_vector_type(8)))  __bf16       v8bf;
typedef __attribute__((ext_vector_type(8)))  float        v8f;
typedef __attribute__((ext_vector_type(4)))  unsigned int v4u;
typedef __attribute__((ext_vector_type(8)))  unsigned int v8u;

__device__ __forceinline__ v16bf cat8(v8bf a, v8bf b) {
  return __builtin_shufflevector(a, b, 0,1,2,3,4,5,6,7,8,9,10,11,12,13,14,15);
}

__device__ __forceinline__ v8f wmma_bf16(v16bf a, v16bf b, v8f c) {
  return __builtin_amdgcn_wmma_f32_16x16x32_bf16(false, a, false, b, (short)0, c,
                                                 false, false);
}

__device__ __forceinline__ v4u shflx16_v4(v4u v) {
  v4u r;
#pragma unroll
  for (int i = 0; i < 4; ++i) r[i] = (unsigned)__shfl_xor((int)v[i], 16, 32);
  return r;
}

// ---------------- elementwise prep kernels ----------------

__global__ void cvt_bf16_kernel(const float* __restrict__ in,
                                __bf16* __restrict__ out, unsigned n) {
  unsigned i = blockIdx.x * 256u + threadIdx.x;
  if (i < n) out[i] = (__bf16)in[i];
}

// in: [K,N] f32 row-major -> out: [N,K] bf16 (transposed); columns n <
// scale_cols are multiplied by scale (folds the softmax 1/sqrt(D)*log2e into
// the q-producing weight columns).
__global__ void transpose_cvt_kernel(const float* __restrict__ in,
                                     __bf16* __restrict__ out, int K, int N,
                                     int scale_cols, float scale) {
  unsigned i = blockIdx.x * 256u + threadIdx.x;
  if (i < (unsigned)(K * N)) {
    int k = i / N;
    int n = i % N;
    float v = in[i];
    if (n < scale_cols) v *= scale;
    out[(size_t)n * K + k] = (__bf16)v;
  }
}

// qkv bf16 [B*T, 3C] (v part) -> VhT [BH, D, T]
__global__ void vtrans_kernel(const __bf16* __restrict__ qkv,
                              __bf16* __restrict__ VhT) {
  constexpr int T = 2048, D = 64, H = 16, C = 1024;
  unsigned i = blockIdx.x * 256u + threadIdx.x;  // over 2*H*T*D = 4194304
  if (i >= 2u * H * T * D) return;
  int d = i & 63;
  int t = (i >> 6) & 2047;
  int h = (i >> 17) & 15;
  int b = i >> 21;
  size_t src = ((size_t)(b * T + t)) * (3 * C) + 2 * C + h * D + d;
  size_t bh = (size_t)b * H + h;
  VhT[(bh * D + d) * T + t] = qkv[src];
}

// ---------------- WMMA GEMM: C[M,N] = A[M,K] * BT[N,K]^T ----------------
// Block = 128 threads = 4 waves arranged 2(m) x 2(n); each wave computes a
// 32(m) x 64(n) tile. Fragments are double-buffered in registers so the
// k+32 loads overlap the 8-deep wmma chain of step k.

template <int STORE_BF16>
__global__ __launch_bounds__(128)
void gemm_bf16_wmma(const __bf16* __restrict__ A,   // [M,K] bf16 row-major
                    const __bf16* __restrict__ BT,  // [N,K] bf16 row-major
                    void* __restrict__ Cout,        // [M,N] bf16 or f32
                    int M, int N, int K) {
  const int lane = threadIdx.x & 31;
  const int wave = threadIdx.x >> 5;
  const int half = lane >> 4;
  const int l16  = lane & 15;
  const int m0 = blockIdx.y * 64 + (wave >> 1) * 32;
  const int n0 = blockIdx.x * 128 + (wave & 1) * 64;

  v8f acc[2][4] = {};
  const __bf16* arow0 = A + (size_t)(m0 + l16) * K + half * 8;
  const __bf16* arow1 = A + (size_t)(m0 + 16 + l16) * K + half * 8;
  const __bf16* bcol  = BT + (size_t)(n0 + l16) * K + half * 8;

  v16bf a0, a1, bb[4];
  a0 = cat8(*(const v8bf*)(arow0), *(const v8bf*)(arow0 + 16));
  a1 = cat8(*(const v8bf*)(arow1), *(const v8bf*)(arow1 + 16));
#pragma unroll
  for (int j = 0; j < 4; ++j) {
    const __bf16* bp = bcol + (size_t)j * 16 * K;
    bb[j] = cat8(*(const v8bf*)(bp), *(const v8bf*)(bp + 16));
  }

  for (int k = 32; k < K; k += 32) {
    v16bf na0 = cat8(*(const v8bf*)(arow0 + k), *(const v8bf*)(arow0 + k + 16));
    v16bf na1 = cat8(*(const v8bf*)(arow1 + k), *(const v8bf*)(arow1 + k + 16));
    v16bf nb[4];
#pragma unroll
    for (int j = 0; j < 4; ++j) {
      const __bf16* bp = bcol + (size_t)j * 16 * K + k;
      nb[j] = cat8(*(const v8bf*)(bp), *(const v8bf*)(bp + 16));
    }
#pragma unroll
    for (int j = 0; j < 4; ++j) {
      acc[0][j] = wmma_bf16(a0, bb[j], acc[0][j]);
      acc[1][j] = wmma_bf16(a1, bb[j], acc[1][j]);
    }
    a0 = na0;
    a1 = na1;
#pragma unroll
    for (int j = 0; j < 4; ++j) bb[j] = nb[j];
  }
#pragma unroll
  for (int j = 0; j < 4; ++j) {
    acc[0][j] = wmma_bf16(a0, bb[j], acc[0][j]);
    acc[1][j] = wmma_bf16(a1, bb[j], acc[1][j]);
  }

#pragma unroll
  for (int mt = 0; mt < 2; ++mt) {
#pragma unroll
    for (int j = 0; j < 4; ++j) {
#pragma unroll
      for (int r = 0; r < 8; ++r) {
        const size_t row = (size_t)(m0 + mt * 16 + half * 8 + r);
        const size_t col = (size_t)(n0 + j * 16 + l16);
        if (STORE_BF16)
          ((__bf16*)Cout)[row * N + col] = (__bf16)acc[mt][j][r];
        else
          ((float*)Cout)[row * N + col] = acc[mt][j][r];
      }
    }
  }
}

// ---------------- sliding-window flash attention (transposed) ----------------
// One wave per (b,h, 16-query tile). Q/K fragments come straight from the qkv
// buffer (row stride 3C, 16B-contiguous per head); V^T is materialized.
// MODE: 0 = no mask, 1 = causal mask + T clamps, 2 = window mask.

__global__ __launch_bounds__(128)
void attn_swin_kernel(const __bf16* __restrict__ qkv,  // [B*T, 3C] bf16
                      const __bf16* __restrict__ VhT,  // [BH, D, T] bf16
                      __bf16* __restrict__ Yb) {       // [B*T, C] bf16
  constexpr int T = 2048, D = 64, H = 16, C = 1024, W = 512;
  constexpr int S3 = 3 * C;  // qkv row stride

  const int lane = threadIdx.x & 31;
  const int wave = threadIdx.x >> 5;
  const int half = lane >> 4;
  const int l16  = lane & 15;

  const int tile = blockIdx.x * 4 + wave;  // [0, 4096)
  const int qt = tile & (T / 16 - 1);
  const int bh = tile >> 7;
  const int h  = bh & (H - 1);
  const int b  = bh >> 4;
  const int q0 = qt * 16;

  const __bf16* Qp = qkv + (size_t)b * T * S3 + h * D;          // + t*S3
  const __bf16* Kp = qkv + (size_t)b * T * S3 + C + h * D;      // + key*S3
  const __bf16* Vp = VhT + (size_t)bh * D * T;

  // Q^T B-fragments (weights pre-scaled): lane col = query q0+l16
  v16bf bq[2];
  {
    const __bf16* qr = Qp + (size_t)(q0 + l16) * S3;
#pragma unroll
    for (int s = 0; s < 2; ++s)
      bq[s] = cat8(*(const v8bf*)(qr + s * 32 + half * 8),
                   *(const v8bf*)(qr + s * 32 + half * 8 + 16));
  }
  // V^T row pointers (A-frag row = d)
  const __bf16* vrow[4];
#pragma unroll
  for (int jj = 0; jj < 4; ++jj) vrow[jj] = Vp + (size_t)(jj * 16 + l16) * T;

  v8f acc[4] = {};   // O^T: rows = d (jj*16 + r + half*8), col = query l16
  float rmax = 0.0f; // per-lane (query l16) running stats, base-2 domain
  float rsum = 0.0f;

  const int i = q0 + l16;  // this lane's query row

  auto do_block = [&](int kk, int mode) {
    // ---- K A-fragments (rows = keys), interleaved score wmmas ----
    int key0 = kk + l16;
    int key1 = kk + 16 + l16;
    if (mode == 1) {  // diagonal block may touch rows >= T
      key0 = key0 < T ? key0 : (T - 1);
      key1 = key1 < T ? key1 : (T - 1);
    }
    const __bf16* kra = Kp + (size_t)key0 * S3 + half * 8;
    const __bf16* krb = Kp + (size_t)key1 * S3 + half * 8;
    v16bf ak00 = cat8(*(const v8bf*)(kra),      *(const v8bf*)(kra + 16));
    v16bf ak01 = cat8(*(const v8bf*)(kra + 32), *(const v8bf*)(kra + 48));
    v16bf ak10 = cat8(*(const v8bf*)(krb),      *(const v8bf*)(krb + 16));
    v16bf ak11 = cat8(*(const v8bf*)(krb + 32), *(const v8bf*)(krb + 48));
    v8f s0 = {}, s1 = {};
    s0 = wmma_bf16(ak00, bq[0], s0);
    s1 = wmma_bf16(ak10, bq[0], s1);
    s0 = wmma_bf16(ak01, bq[1], s0);
    s1 = wmma_bf16(ak11, bq[1], s1);

    // ---- V A-fragments issued early: loads overlap the softmax VALU ----
    const int c0 = kk + half * 8;  // always <= T-8 within the loop ranges
    int c1 = kk + 16 + half * 8;
    if (mode == 1 && c1 > T - 8) c1 = T - 8;  // OOB keys have P==0
    v16bf av[4];
#pragma unroll
    for (int jj = 0; jj < 4; ++jj)
      av[jj] = cat8(*(const v8bf*)(vrow[jj] + c0), *(const v8bf*)(vrow[jj] + c1));

    // ---- per-lane masked online softmax (base-2 domain) ----
    float sv0[8], sv1[8];
    float pm = -3.0e38f;
#pragma unroll
    for (int r = 0; r < 8; ++r) {
      float a = s0[r];
      float bv = s1[r];
      if (mode == 1) {  // causal only (window bound trivially satisfied)
        const int j0 = kk + half * 8 + r;
        const int j1 = kk + 16 + half * 8 + r;
        a  = (j0 <= i) ? a  : -3.0e38f;
        bv = (j1 <= i) ? bv : -3.0e38f;
      } else if (mode == 2) {  // window only (j <= i trivially satisfied)
        const int j0 = kk + half * 8 + r;
        const int j1 = kk + 16 + half * 8 + r;
        a  = (j0 > i - W) ? a  : -3.0e38f;
        bv = (j1 > i - W) ? bv : -3.0e38f;
      }
      sv0[r] = a;
      sv1[r] = bv;
      pm = fmaxf(pm, fmaxf(a, bv));
    }
    float bm = fmaxf(pm, __shfl_xor(pm, 16, 32));  // partner holds other 16 keys
    const float nm  = fmaxf(rmax, bm);
    const float fac = __builtin_amdgcn_exp2f(rmax - nm);
    v8bf e0, e1;
    float ps = 0.0f;
#pragma unroll
    for (int r = 0; r < 8; ++r) {
      float p0 = __builtin_amdgcn_exp2f(sv0[r] - nm);
      float p1 = __builtin_amdgcn_exp2f(sv1[r] - nm);
      ps += p0 + p1;
      e0[r] = (__bf16)p0;
      e1[r] = (__bf16)p1;
    }
    ps += __shfl_xor(ps, 16, 32);
    rsum = rsum * fac + ps;
    rmax = nm;
#pragma unroll
    for (int jj = 0; jj < 4; ++jj)
#pragma unroll
      for (int r = 0; r < 8; ++r) acc[jj][r] *= fac;

    // ---- P^T B-frag: elements = keys kk + half*8 + i, kk + 16 + half*8 + i ----
    v4u se0 = __builtin_bit_cast(v4u, e0);
    v4u se1 = __builtin_bit_cast(v4u, e1);
    v4u oe0 = shflx16_v4(se0);
    v4u oe1 = shflx16_v4(se1);
    v4u lo = se0;
    v4u hi = se1;
    v8u pcat;
#pragma unroll
    for (int c = 0; c < 4; ++c) { pcat[c] = lo[c]; pcat[4 + c] = hi[c]; }
    v16bf pf = __builtin_bit_cast(v16bf, pcat);

    // ---- O^T += V^T * P^T (4 independent wmma chains) ----
#pragma unroll
    for (int jj = 0; jj < 4; ++jj) acc[jj] = wmma_bf16(av[jj], pf, acc[jj]);
  };

  // kk sequence: kk_begin, kk_begin+32, ..., diag_kk
  const int kk_begin = (qt >= 32) ? (q0 - 512) : 0;
  const int diag_kk  = kk_begin + (((q0 - kk_begin) >> 5) << 5);

  int kk = kk_begin;
  if (qt >= 32) {           // window-edge block: window mask only
    do_block(kk, 2);
    kk += 32;
  }
  for (; kk < diag_kk; kk += 32) do_block(kk, 0);  // unmasked body
  do_block(diag_kk, 1);     // diagonal block: causal mask + clamps

  // ---- normalize (per-lane query) and store y: 16B vector stores ----
  const float inv = 1.0f / rsum;
  __bf16* yrow = Yb + (size_t)(b * T + q0 + l16) * C + h * D;
#pragma unroll
  for (int jj = 0; jj < 4; ++jj) {
    v8bf o;
#pragma unroll
    for (int r = 0; r < 8; ++r) o[r] = (__bf16)(acc[jj][r] * inv);
    *(v8bf*)(yrow + jj * 16 + half * 8) = o;
  }
}

// ---------------- launcher ----------------

extern "C" void kernel_launch(void* const* d_in, const int* in_sizes, int n_in,
                              void* d_out, int out_size, void* d_ws, size_t ws_size,
                              hipStream_t stream) {
  (void)in_sizes; (void)n_in; (void)out_size; (void)ws_size;
  const float* x      = (const float*)d_in[0];  // [2,2048,1024]
  const float* w_attn = (const float*)d_in[1];  // [1024,3072]
  const float* w_proj = (const float*)d_in[2];  // [1024,1024]
  float* out = (float*)d_out;                   // [2,2048,1024] f32

  constexpr int B = 2, T = 2048, C = 1024, H = 16, D = 64;
  constexpr int M = B * T;          // 4096
  constexpr int N1 = 3 * C;         // 3072
  constexpr float QSCALE = 0.18033688011112042f;  // (1/sqrt(64)) * log2(e)

  char* ws = (char*)d_ws;
  __bf16* xb   = (__bf16*)(ws);                          //  8.0 MiB [4096,1024]
  __bf16* wAT  = (__bf16*)(ws + 8388608);                //  6.0 MiB [3072,1024]
  __bf16* wPT  = (__bf16*)(ws + 14680064);               //  2.0 MiB [1024,1024]
  __bf16* qkvb = (__bf16*)(ws + 16777216);               // 24.0 MiB [4096,3072]
  __bf16* VhT  = (__bf16*)(ws + 41943040);               //  8.0 MiB [32,64,2048]
  __bf16* Yb   = (__bf16*)(ws + 50331648);               //  8.0 MiB [4096,1024]

  // 1) precision / layout prep (q-scale folded into w_attn's first C columns)
  cvt_bf16_kernel<<<(M * C) / 256, 256, 0, stream>>>(x, xb, M * C);
  transpose_cvt_kernel<<<(C * N1) / 256, 256, 0, stream>>>(w_attn, wAT, C, N1,
                                                           C, QSCALE);
  transpose_cvt_kernel<<<(C * C) / 256, 256, 0, stream>>>(w_proj, wPT, C, C,
                                                          0, 1.0f);

  // 2) qkv = x @ w_attn   (bf16 out; q columns pre-scaled)
  gemm_bf16_wmma<1><<<dim3(N1 / 128, M / 64), 128, 0, stream>>>(xb, wAT, qkvb,
                                                                M, N1, C);
  // 3) V^T only (Q/K are consumed in place from qkvb)
  vtrans_kernel<<<(B * H * T * D) / 256, 256, 0, stream>>>(qkvb, VhT);

  // 4) sliding-window attention -> Yb
  attn_swin_kernel<<<(B * H * (T / 16)) / 4, 128, 0, stream>>>(qkvb, VhT, Yb);

  // 5) out = y @ w_proj   (f32 out)
  gemm_bf16_wmma<0><<<dim3(C / 128, M / 64), 128, 0, stream>>>(Yb, wPT, out,
                                                               M, C, C);
}
